// Speller_87299505258600
// MI455X (gfx1250) — matrix operand order repaired and checked by reading, not compiled, not run on hardware
//
#include <hip/hip_runtime.h>

// ---------------------------------------------------------------------------
// Speller (LAS decoder): B=128, T=500, L=250, H=512, E=256, V=34
//   * one-time: enc_key/enc_value -> bf16 (131 MB total -> L2-resident),
//               weights -> bf16 pre-packed in WMMA B-fragment layout
//   * per step: lstm1 (WMMA bf16, 4 gates fused, software-pipelined)
//               -> lstm2 -> attention (+ next-step emb gather) -> logits
//   * activations assembled in-place into double-buffered matrices:
//       X1[2][128][1280] = [emb_t | ctx | h0]   (lstm1 A operand)
//       X2[2][128][1024] = [h0 | h1]            (lstm2 A operand)
//     so the GEMM inner loop is branch-free contiguous b128 loads.
// ---------------------------------------------------------------------------

typedef __attribute__((ext_vector_type(16))) __bf16   v16bf;
typedef __attribute__((ext_vector_type(8)))  float    v8f;
typedef __attribute__((ext_vector_type(4)))  unsigned u32x4;

union ABFrag { u32x4 u[2]; v16bf v; };
struct Frags { ABFrag a; ABFrag b[4]; };

#define Bn   128
#define Tn   500
#define Ln   250
#define Hn   512
#define En   256
#define Vn   34
#define BH   (Bn * Hn)            // 65536
#define K1n  1280                 // emb(256) | ctx(512) | h0(512)
#define K2n  1024                 // h0(512) | h1(512)

__device__ __forceinline__ float bflo(unsigned u) { return __uint_as_float(u << 16); }
__device__ __forceinline__ float bfhi(unsigned u) { return __uint_as_float(u & 0xffff0000u); }
__device__ __forceinline__ float sigf(float x)    { return 1.f / (1.f + __expf(-x)); }

// ---------------- one-time preprocessing kernels ----------------

__global__ void cvt_bf16_kernel(const float* __restrict__ s, __bf16* __restrict__ d, long n) {
    long i = (long)blockIdx.x * blockDim.x + threadIdx.x;
    long stride = (long)gridDim.x * blockDim.x;
    for (; i < n; i += stride) d[i] = (__bf16)s[i];
}

// Pack W^T = [W_ih ; W_hh]^T (K x 4H) into WMMA B-fragment layout:
//   off = ((nb*(K/32) + kc)*32 + lane)*16 + i
//   lane = ((kr>>3)&1)*16 + (n&15),  i = (kr&7) | ((kr>>4)<<3),  kc = k>>5
__global__ void pack_w_kernel(const float* __restrict__ Wih, const float* __restrict__ Whh,
                              int Kih, int K, __bf16* __restrict__ Wp) {
    long idx = (long)blockIdx.x * blockDim.x + threadIdx.x;  // over K*2048
    if (idx >= (long)K * 2048) return;
    int n = (int)(idx & 2047);
    int k = (int)(idx >> 11);
    int Khh = K - Kih;
    float w = (k < Kih) ? Wih[(size_t)n * Kih + k] : Whh[(size_t)n * Khh + (k - Kih)];
    int nb = n >> 4, ln = n & 15;
    int kc = k >> 5, kr = k & 31;
    int hi = (kr >> 3) & 1, i = (kr & 7) | ((kr >> 4) << 3);
    int lane = hi * 16 + ln;
    size_t off = (((size_t)nb * (K >> 5) + kc) * 32 + lane) * 16 + i;
    Wp[off] = (__bf16)w;
}

// Initialize the parity the first step reads: X1[1] = [emb_0 | enc_key[:,0,:] | 0],
// X2[0] = [0 | 0], c0 = c1 = 0.
__global__ void init_state_kernel(const float* __restrict__ enc_key,
                                  const float* __restrict__ emb, const int* __restrict__ decode,
                                  __bf16* __restrict__ X1, __bf16* __restrict__ X2,
                                  float* __restrict__ c0f, float* __restrict__ c1f) {
    int idx = blockIdx.x * blockDim.x + threadIdx.x;  // over B*H
    if (idx >= BH) return;
    int b = idx >> 9, h = idx & (Hn - 1);
    __bf16 z = (__bf16)0.f;
    __bf16* x1 = X1 + (size_t)(Bn + b) * K1n;   // parity 1
    x1[En + h]      = (__bf16)enc_key[(size_t)b * Tn * Hn + h];  // ctx_init = enc_key[:,0,:]
    x1[En + Hn + h] = z;                                         // h0 = 0
    __bf16* x2 = X2 + (size_t)b * K2n;          // parity 0
    x2[h] = z; x2[Hn + h] = z;                  // h0 = h1 = 0
    c0f[idx] = 0.f; c1f[idx] = 0.f;
    if (h < En) {
        int ch = decode[b * Ln];                // emb_0
        x1[h] = (__bf16)emb[(size_t)ch * En + h];
    }
}

// ---------------- fused LSTM cell (WMMA bf16, software-pipelined) ----------------
// g = A @ W^T + b_ih + b_hh ; i,f,g,o gates ; c' = f*c + i*tanh(g); h = o*tanh(c')
// 256 threads = 8 waves; wave tile = 16 batch x 16 hidden, 4 gate accumulators.
__device__ __forceinline__ void load_frags(Frags& f, const __bf16* arow,
                                           const __bf16* const* bbase, int kc) {
    const __bf16* ap = arow + kc * 32;
    f.a.u[0] = *(const u32x4*)(ap);
    f.a.u[1] = *(const u32x4*)(ap + 16);
#pragma unroll
    for (int g = 0; g < 4; ++g) {
        const __bf16* bp = bbase[g] + (size_t)kc * 512;
        f.b[g].u[0] = *(const u32x4*)(bp);
        f.b[g].u[1] = *(const u32x4*)(bp + 8);
    }
}

__global__ __launch_bounds__(256) void lstm_kernel(
    const __bf16* __restrict__ A, int Ktot,
    const __bf16* __restrict__ Wp,
    const float* __restrict__ bih, const float* __restrict__ bhh,
    float* __restrict__ c_state,
    __bf16* __restrict__ h_out, int ldh,
    __bf16* __restrict__ h_out2, int ldh2)   // h_out2 may be null
{
    int wave = threadIdx.x >> 5;
    int lane = threadIdx.x & 31;
    int tile = blockIdx.x * 8 + wave;     // 0..255
    int mt = tile >> 5;                   // batch tile  0..7
    int nt = tile & 31;                   // hidden tile 0..31
    int hi = lane >> 4;
    int lm = lane & 15;

    v8f acc[4];
    v8f zero = {0.f, 0.f, 0.f, 0.f, 0.f, 0.f, 0.f, 0.f};
    acc[0] = zero; acc[1] = zero; acc[2] = zero; acc[3] = zero;

    const int nchunks = Ktot >> 5;
    const size_t chunkstride = (size_t)nchunks * 512;   // elements per 16-col block of Wp
    const int mrow = mt * 16 + lm;

    // A fragment rows: lane holds K = {kb+hi*8 .. +7} and {kb+16+hi*8 .. +7}
    const __bf16* arow = A + (size_t)mrow * Ktot + hi * 8;
    const __bf16* bbase[4];
#pragma unroll
    for (int g = 0; g < 4; ++g)
        bbase[g] = Wp + (size_t)(g * 32 + nt) * chunkstride + (size_t)lane * 16;

    // software pipeline: loads for chunk kc+1 issued before WMMAs of chunk kc
    Frags cur, nxt;
    load_frags(cur, arow, bbase, 0);
    for (int kc = 0; kc < nchunks - 1; ++kc) {
        load_frags(nxt, arow, bbase, kc + 1);
#pragma unroll
        for (int g = 0; g < 4; ++g)
            acc[g] = __builtin_amdgcn_wmma_f32_16x16x32_bf16(
                false, cur.a.v, false, cur.b[g].v, (short)0, acc[g], false, false);
        cur = nxt;
    }
#pragma unroll
    for (int g = 0; g < 4; ++g)
        acc[g] = __builtin_amdgcn_wmma_f32_16x16x32_bf16(
            false, cur.a.v, false, cur.b[g].v, (short)0, acc[g], false, false);

    int j = nt * 16 + lm;
    float bi = bih[j]          + bhh[j];
    float bf = bih[Hn + j]     + bhh[Hn + j];
    float bg = bih[2 * Hn + j] + bhh[2 * Hn + j];
    float bo = bih[3 * Hn + j] + bhh[3 * Hn + j];

#pragma unroll
    for (int r = 0; r < 8; ++r) {
        int m = mt * 16 + hi * 8 + r;
        float gi = acc[0][r] + bi;
        float gf = acc[1][r] + bf;
        float gg = acc[2][r] + bg;
        float go = acc[3][r] + bo;
        float is = sigf(gi), fs = sigf(gf), os = sigf(go);
        float gt = tanhf(gg);
        size_t cidx = (size_t)m * Hn + j;
        float cn = fs * c_state[cidx] + is * gt;
        c_state[cidx] = cn;
        __bf16 hb = (__bf16)(os * tanhf(cn));
        h_out[(size_t)m * ldh + j] = hb;
        if (h_out2) h_out2[(size_t)m * ldh2 + j] = hb;
    }
}

// ---------------- attention: energy -> masked softmax -> context ----------------
// one workgroup (256 threads) per batch row; bf16 enc tensors (L2-resident).
// Also gathers the NEXT step's char embedding into the X1 write buffer.
__global__ __launch_bounds__(256) void attn_kernel(
    const __bf16* __restrict__ ekb, const __bf16* __restrict__ evb,
    const __bf16* __restrict__ h1, int ldh,
    const int* __restrict__ slens,
    __bf16* __restrict__ ctx_out, int ldc,
    const float* __restrict__ emb, const int* __restrict__ decode, int tnext,
    __bf16* __restrict__ embdst, int lde)
{
    __shared__ float hbuf[Hn];
    __shared__ float ebuf[Tn];
    __shared__ float red[8];

    int b = blockIdx.x, tid = threadIdx.x;

    // gather emb_{t+1} for next step's lstm1 A-matrix (256 threads == En)
    if (tnext >= 0) {
        int ch = decode[b * Ln + tnext];
        embdst[(size_t)b * lde + tid] = (__bf16)emb[(size_t)ch * En + tid];
    }

    for (int h = tid; h < Hn; h += 256) hbuf[h] = (float)h1[(size_t)b * ldh + h];
    __syncthreads();

    int slen = slens[b];
    for (int t = tid; t < Tn; t += 256) {
        const __bf16* row = ekb + ((size_t)b * Tn + t) * Hn;
        float s = 0.f;
        for (int h = 0; h < Hn; h += 8) {
            u32x4 u = *(const u32x4*)(row + h);
            s += hbuf[h + 0] * bflo(u[0]) + hbuf[h + 1] * bfhi(u[0]);
            s += hbuf[h + 2] * bflo(u[1]) + hbuf[h + 3] * bfhi(u[1]);
            s += hbuf[h + 4] * bflo(u[2]) + hbuf[h + 5] * bfhi(u[2]);
            s += hbuf[h + 6] * bflo(u[3]) + hbuf[h + 7] * bfhi(u[3]);
        }
        ebuf[t] = (t >= slen) ? -1e-9f : s;   // reference uses -1e-09 (sic)
    }
    __syncthreads();

    // max
    float m = -3.4e38f;
    for (int t = tid; t < Tn; t += 256) m = fmaxf(m, ebuf[t]);
    for (int o = 16; o; o >>= 1) m = fmaxf(m, __shfl_xor(m, o, 32));
    if ((tid & 31) == 0) red[tid >> 5] = m;
    __syncthreads();
    float M = red[0];
#pragma unroll
    for (int i = 1; i < 8; ++i) M = fmaxf(M, red[i]);
    __syncthreads();

    // exp + sum
    float ssum = 0.f;
    for (int t = tid; t < Tn; t += 256) {
        float e = __expf(ebuf[t] - M);
        ebuf[t] = e;
        ssum += e;
    }
    for (int o = 16; o; o >>= 1) ssum += __shfl_xor(ssum, o, 32);
    if ((tid & 31) == 0) red[tid >> 5] = ssum;
    __syncthreads();
    float tot = 0.f;
#pragma unroll
    for (int i = 0; i < 8; ++i) tot += red[i];
    float inv = 1.f / tot;

    // ctx[h] = sum_t attn[t] * ev[b][t][h]; each thread owns 2 adjacent h (one u32 load)
    int h2 = tid * 2;
    const __bf16* evrow = evb + (size_t)b * Tn * Hn + h2;
    float a0 = 0.f, a1 = 0.f;
    for (int t = 0; t < Tn; ++t) {
        float a = ebuf[t];
        unsigned u = *(const unsigned*)(evrow + (size_t)t * Hn);
        a0 += a * bflo(u);
        a1 += a * bfhi(u);
    }
    ctx_out[(size_t)b * ldc + h2]     = (__bf16)(a0 * inv);
    ctx_out[(size_t)b * ldc + h2 + 1] = (__bf16)(a1 * inv);
}

// ---------------- output projection (tiny: V=34) ----------------
__global__ __launch_bounds__(64) void logits_kernel(
    const __bf16* __restrict__ h1, int ldh,
    const __bf16* __restrict__ ctx, int ldc,
    const float* __restrict__ Wout, const float* __restrict__ bout,
    float* __restrict__ out, int t)
{
    __shared__ float xb[2 * Hn];
    int b = blockIdx.x, tid = threadIdx.x;
    for (int i = tid; i < Hn; i += 64) {
        xb[i]      = (float)h1[(size_t)b * ldh + i];
        xb[Hn + i] = (float)ctx[(size_t)b * ldc + i];
    }
    __syncthreads();
    if (tid < Vn) {
        const float* w = Wout + (size_t)tid * (2 * Hn);
        float s = bout[tid];
        for (int k = 0; k < 2 * Hn; ++k) s += w[k] * xb[k];
        out[((size_t)b * Ln + t) * Vn + tid] = s;
    }
}

// ---------------- host driver ----------------

extern "C" void kernel_launch(void* const* d_in, const int* in_sizes, int n_in,
                              void* d_out, int out_size, void* d_ws, size_t ws_size,
                              hipStream_t stream) {
    const float* enc_key   = (const float*)d_in[0];
    const float* enc_value = (const float*)d_in[1];
    const float* emb       = (const float*)d_in[2];
    const float* W_ih1     = (const float*)d_in[3];
    const float* W_hh1     = (const float*)d_in[4];
    const float* b_ih1     = (const float*)d_in[5];
    const float* b_hh1     = (const float*)d_in[6];
    const float* W_ih2     = (const float*)d_in[7];
    const float* W_hh2     = (const float*)d_in[8];
    const float* b_ih2     = (const float*)d_in[9];
    const float* b_hh2     = (const float*)d_in[10];
    const float* W_out     = (const float*)d_in[11];
    const float* b_out     = (const float*)d_in[12];
    const int*   slens     = (const int*)d_in[13];
    const int*   decode    = (const int*)d_in[14];
    float* out = (float*)d_out;

    char* cur = (char*)d_ws;
    auto carve = [&](size_t bytes) -> char* {
        char* p = cur;
        cur += (bytes + 255) & ~(size_t)255;
        return p;
    };
    __bf16* ekb = (__bf16*)carve((size_t)Bn * Tn * Hn * 2);   // 65.5 MB
    __bf16* evb = (__bf16*)carve((size_t)Bn * Tn * Hn * 2);   // 65.5 MB
    __bf16* W1p = (__bf16*)carve((size_t)K1n * 2048 * 2);     //  5.2 MB
    __bf16* W2p = (__bf16*)carve((size_t)K2n * 2048 * 2);     //  4.2 MB
    __bf16* X1  = (__bf16*)carve((size_t)2 * Bn * K1n * 2);   //  0.66 MB
    __bf16* X2  = (__bf16*)carve((size_t)2 * Bn * K2n * 2);   //  0.52 MB
    float*  c0f = (float*)carve((size_t)BH * 4);
    float*  c1f = (float*)carve((size_t)BH * 4);

    // one-time preprocessing (re-done every call: deterministic)
    long nenc = (long)Bn * Tn * Hn;
    cvt_bf16_kernel<<<8192, 256, 0, stream>>>(enc_key, ekb, nenc);
    cvt_bf16_kernel<<<8192, 256, 0, stream>>>(enc_value, evb, nenc);
    pack_w_kernel<<<(K1n * 2048 + 255) / 256, 256, 0, stream>>>(W_ih1, W_hh1, En + Hn, K1n, W1p);
    pack_w_kernel<<<(K2n * 2048 + 255) / 256, 256, 0, stream>>>(W_ih2, W_hh2, Hn, K2n, W2p);
    init_state_kernel<<<(BH + 255) / 256, 256, 0, stream>>>(enc_key, emb, decode, X1, X2, c0f, c1f);

    for (int t = 0; t < Ln; ++t) {
        int q = t & 1;
        __bf16* X1r = X1 + (size_t)(q ^ 1) * Bn * K1n;  // [emb_t | ctx_{t-1} | h0_{t-1}]
        __bf16* X1w = X1 + (size_t)q       * Bn * K1n;  // assembled for step t+1
        __bf16* X2t = X2 + (size_t)q       * Bn * K2n;  // [h0_t | h1_{t-1}] (lstm2 input)
        __bf16* X2n = X2 + (size_t)(q ^ 1) * Bn * K2n;  // receives h1_t

        // LSTM1: A = X1r (K=1280); h0_t -> X2t[:,0:512] and X1w[:,768:1280]
        lstm_kernel<<<32, 256, 0, stream>>>(X1r, K1n, W1p, b_ih1, b_hh1, c0f,
                                            X2t, K2n, X1w + En + Hn, K1n);
        // LSTM2: A = X2t (K=1024); h1_t -> X2n[:,512:1024]
        lstm_kernel<<<32, 256, 0, stream>>>(X2t, K2n, W2p, b_ih2, b_hh2, c1f,
                                            X2n + Hn, K2n, nullptr, 0);
        // attention: ctx_t -> X1w[:,256:768]; also gathers emb_{t+1} -> X1w[:,0:256]
        attn_kernel<<<Bn, 256, 0, stream>>>(ekb, evb, X2n + Hn, K2n, slens,
                                            X1w + En, K1n,
                                            emb, decode, (t + 1 < Ln) ? (t + 1) : -1,
                                            X1w, K1n);
        // logits
        logits_kernel<<<Bn, 64, 0, stream>>>(X2n + Hn, K2n, X1w + En, K1n,
                                             W_out, b_out, out, t);
    }
}
